// SwitchSAE_13185549598759
// MI455X (gfx1250) — matrix operand order repaired and checked by reading, not compiled, not run on hardware
//
#include <hip/hip_runtime.h>
#include <hip/hip_bf16.h>

#define DDIM 768
#define LDIM 3072
#define NEXP 8
#define KTOP 32
#define BTOK 4096
#define KSTEPS (DDIM / 32)     // 24
#define MTOK 32                // tokens per encode block
#define MTMAX (BTOK / MTOK)    // 128 tiles per expert worst case
#define NLAT 256               // latents per encode block (4 waves x 64)
#define XBPAD 40               // bf16 elems per (kstep,lane) row: 32 used + pad (80B, 16B-aligned)

typedef __attribute__((ext_vector_type(16))) __bf16 v16bf;
typedef __attribute__((ext_vector_type(8)))  float  v8f;

// ------------------------------------------------- W_enc f32 -> bf16
__global__ __launch_bounds__(256) void convert_wenc_kernel(
    const float* __restrict__ w, __bf16* __restrict__ wb)
{
    const size_t i = ((size_t)blockIdx.x * 256 + threadIdx.x) * 8;
    const float4 f0 = *(const float4*)(w + i);
    const float4 f1 = *(const float4*)(w + i + 4);
    union { __bf16 h[8]; uint4 q; } u;
    u.h[0] = (__bf16)f0.x; u.h[1] = (__bf16)f0.y;
    u.h[2] = (__bf16)f0.z; u.h[3] = (__bf16)f0.w;
    u.h[4] = (__bf16)f1.x; u.h[5] = (__bf16)f1.y;
    u.h[6] = (__bf16)f1.z; u.h[7] = (__bf16)f1.w;
    *(uint4*)(wb + i) = u.q;
}

// ---------------------------------------------------------------- router
__global__ __launch_bounds__(256) void router_kernel(
    const float* __restrict__ x, const float* __restrict__ rw,
    const float* __restrict__ rb,
    int* __restrict__ counts, float* __restrict__ sumsq,
    float* __restrict__ sum_x, float* __restrict__ sel_probs,
    int* __restrict__ eids, float* __restrict__ eid_out)
{
    const int wave = threadIdx.x >> 5;
    const int lane = threadIdx.x & 31;
    const int t = blockIdx.x * 8 + wave;
    if (t >= BTOK) return;
    const float* xr = x + (size_t)t * DDIM;

    float xl[DDIM / 32];
#pragma unroll
    for (int j = 0; j < DDIM / 32; ++j) xl[j] = xr[lane + 32 * j];

    float logit[NEXP];
#pragma unroll
    for (int e = 0; e < NEXP; ++e) {
        const float* w = rw + e * DDIM;
        float s = 0.f;
#pragma unroll
        for (int j = 0; j < DDIM / 32; ++j) s += xl[j] * w[lane + 32 * j];
#pragma unroll
        for (int off = 16; off > 0; off >>= 1) s += __shfl_xor(s, off, 32);
        logit[e] = s + rb[e];
    }
    float m = logit[0]; int am = 0;
#pragma unroll
    for (int e = 1; e < NEXP; ++e)
        if (logit[e] > m) { m = logit[e]; am = e; }
    float den = 0.f;
#pragma unroll
    for (int e = 0; e < NEXP; ++e) den += __expf(logit[e] - m);
    const float sp = 1.0f / den;

    float ss = 0.f;
#pragma unroll
    for (int j = 0; j < DDIM / 32; ++j) ss += xl[j] * xl[j];
#pragma unroll
    for (int off = 16; off > 0; off >>= 1) ss += __shfl_xor(ss, off, 32);

    if (lane == 0) {
        eids[t] = am;
        sel_probs[t] = sp;
        eid_out[t] = (float)am;
        atomicAdd(&counts[am], 1);
        atomicAdd(&sumsq[am], ss);
    }
    float* sx = sum_x + am * DDIM;
#pragma unroll
    for (int j = 0; j < DDIM / 32; ++j) atomicAdd(&sx[lane + 32 * j], xl[j]);
}

// ------------------------------------------------------------- offsets
__global__ void offsets_kernel(const int* __restrict__ counts,
                               int* __restrict__ offsets,
                               int* __restrict__ cursor)
{
    if (threadIdx.x == 0) {
        int acc = 0;
        for (int e = 0; e < NEXP; ++e) {
            offsets[e] = acc; cursor[e] = acc; acc += counts[e];
        }
        offsets[NEXP] = acc;
    }
}

// ------------------------------------------------------------- scatter
__global__ __launch_bounds__(256) void scatter_kernel(
    const int* __restrict__ eids, int* __restrict__ cursor,
    int* __restrict__ tok_list)
{
    const int t = blockIdx.x * 256 + threadIdx.x;
    if (t < BTOK) {
        const int p = atomicAdd(&cursor[eids[t]], 1);
        tok_list[p] = t;
    }
}

// ------------------------------------------------- encode GEMM (WMMA bf16)
// Block: 128 thr = 4 waves; tile: 32 tokens x 256 latents (64 latents/wave).
// A = pre-converted bf16 W_enc: 2x global b128 per fragment, no cvt.
// B = x tile pre-swizzled in LDS: 2x ds_load_b128 per fragment.
// 8 WMMAs per wave per k-step; branch-free inner loop.
__global__ __launch_bounds__(128) void encode_kernel(
    const float* __restrict__ x, const __bf16* __restrict__ Wencb,
    const float* __restrict__ b_enc, const float* __restrict__ b_dec,
    const int* __restrict__ offsets, const int* __restrict__ tok_list,
    float* __restrict__ pre)
{
    __shared__ __align__(16) __bf16 XB[KSTEPS][32][XBPAD]; // 60 KB
    __shared__ float Dout[MTOK][NLAT + 5];                 // 32x261 f32, 33.4 KB
    __shared__ int   toks[MTOK];

    const int e  = blockIdx.y >> 7;            // blockIdx.y = e*MTMAX + mt
    const int mt = blockIdx.y & (MTMAX - 1);
    const int seg0 = offsets[e], seg1 = offsets[e + 1];
    const int m0 = seg0 + mt * MTOK;
    if (m0 >= seg1) return;                    // uniform early exit

    const int n0  = blockIdx.x * NLAT;
    const int tid = threadIdx.x;

    if (tid < MTOK)
        toks[tid] = (m0 + tid < seg1) ? tok_list[m0 + tid] : -1;
    __syncthreads();

    // ---- stage x tile into B-fragment layout: XB[kb][(i&15)+16h][(i>>4)*16+s] = x[tok_i][kb*32+k(h,s)] - b_dec (table k order)
    for (int slot = tid; slot < KSTEPS * 32; slot += 128) {
        const int kb = slot >> 5, l = slot & 31;   // col = slot
        const float bd = b_dec[e * DDIM + slot];
        const int hh = (l >> 3) & 1, ss = (l & 7) + ((l >> 4) << 3);
#pragma unroll
        for (int i = 0; i < MTOK; ++i) {
            const int tk = toks[i];
            const float v = (tk >= 0) ? (x[(size_t)tk * DDIM + slot] - bd) : 0.f;
            XB[kb][(i & 15) + 16 * hh][((i >> 4) << 4) + ss] = (__bf16)v;
        }
    }
    __syncthreads();

    const int wave = tid >> 5, lane = tid & 31;
    const int nW0 = n0 + wave * 64;                    // this wave's 64 latents
    const __bf16* Wb = Wencb + (size_t)e * LDIM * DDIM;
    const int lrow  = lane & 15;
    const int khalf = (lane < 16) ? 0 : 8;

    const __bf16* aptr[4];
#pragma unroll
    for (int lt = 0; lt < 4; ++lt)
        aptr[lt] = Wb + (size_t)(nW0 + lt * 16 + lrow) * DDIM + khalf;
    const __bf16* xbl = &XB[0][lane][0];

    union VB { uint4 q[2]; v16bf v; };

    v8f c0[4] = {}, c1[4] = {};       // c0: token tile 0, c1: token tile 1
    for (int kb = 0; kb < KSTEPS; ++kb) {
        const int k0 = kb * 32;
        // unconditional speculative prefetch of next k-slice (no loop split)
        __builtin_prefetch(aptr[0] + k0 + 32, 0, 1);
        __builtin_prefetch(aptr[2] + k0 + 32, 0, 1);

        // ---- B fragments: contiguous 32B per lane from pre-swizzled LDS
        const __bf16* bp = xbl + (size_t)kb * (32 * XBPAD);
        VB b0u, b1u;
        b0u.q[0] = *(const uint4*)(bp + 0);
        b0u.q[1] = *(const uint4*)(bp + 8);
        b1u.q[0] = *(const uint4*)(bp + 16);
        b1u.q[1] = *(const uint4*)(bp + 24);

        // ---- all 4 A fragments first (one clause of 8x b128), then 8 WMMAs
        VB au[4];
#pragma unroll
        for (int lt = 0; lt < 4; ++lt) {
            au[lt].q[0] = *(const uint4*)(aptr[lt] + k0);        // K = k0+khalf .. +7
            au[lt].q[1] = *(const uint4*)(aptr[lt] + k0 + 16);   // K = k0+16+khalf .. +7
        }
#pragma unroll
        for (int lt = 0; lt < 4; ++lt) {
            c0[lt] = __builtin_amdgcn_wmma_f32_16x16x32_bf16(
                         false, au[lt].v, false, b0u.v, (short)0, c0[lt], false, false);
            c1[lt] = __builtin_amdgcn_wmma_f32_16x16x32_bf16(
                         false, au[lt].v, false, b1u.v, (short)0, c1[lt], false, false);
        }
    }

    // ---- D layout: M(latent) = r + 8*(lane>=16), N(token) = lane&15
    {
        const int tokN  = lane & 15;
        const int mhalf = (lane < 16) ? 0 : 8;
#pragma unroll
        for (int lt = 0; lt < 4; ++lt) {
#pragma unroll
            for (int r = 0; r < 8; ++r) {
                const int ll = wave * 64 + lt * 16 + r + mhalf;
                Dout[tokN][ll]      = c0[lt][r];
                Dout[tokN + 16][ll] = c1[lt][r];
            }
        }
    }
    __syncthreads();

    // ---- bias + ReLU + coalesced store to pre
    const float* be = b_enc + e * LDIM + n0;
#pragma unroll
    for (int g = 0; g < 4; ++g) {
        const int token = (tid >> 4) + 8 * g;      // 8 token-groups of 16 threads
        const int gm = m0 + token;
        if (gm < seg1) {
            float* dst = pre + (size_t)toks[token] * LDIM + n0;
            for (int col = (tid & 15); col < NLAT; col += 16) {
                const float v = Dout[token][col] + be[col];
                dst[col] = v > 0.f ? v : 0.f;
            }
        }
    }
}

// ------------------------------------------------------------- top-k
__global__ __launch_bounds__(256) void topk_kernel(
    const float* __restrict__ pre, float* __restrict__ acts_out,
    float* __restrict__ idx_out)
{
    __shared__ float sh[LDIM];
    __shared__ float rv[256];
    __shared__ int   ri[256];
    const int t = blockIdx.x, tid = threadIdx.x;
    const float* row = pre + (size_t)t * LDIM;
    for (int i = tid; i < LDIM; i += 256) sh[i] = row[i];
    __syncthreads();

    for (int k = 0; k < KTOP; ++k) {
        float bv = -1e30f; int bi = LDIM;
#pragma unroll
        for (int j = 0; j < LDIM / 256; ++j) {
            const int i = tid + j * 256;
            const float v = sh[i];
            if (v > bv) { bv = v; bi = i; }   // strict > keeps lowest index
        }
        rv[tid] = bv; ri[tid] = bi;
        __syncthreads();
        for (int s = 128; s > 0; s >>= 1) {
            if (tid < s) {
                const float ov = rv[tid + s]; const int oi = ri[tid + s];
                if (ov > rv[tid] || (ov == rv[tid] && oi < ri[tid])) {
                    rv[tid] = ov; ri[tid] = oi;
                }
            }
            __syncthreads();
        }
        if (tid == 0) {
            acts_out[t * KTOP + k] = rv[0];
            idx_out[t * KTOP + k]  = (float)ri[0];
            sh[ri[0]] = -1e30f;
        }
        __syncthreads();
    }
}

// ------------------------------------------------------------- decode
__global__ __launch_bounds__(256) void decode_kernel(
    const float* __restrict__ x, const float* __restrict__ W_dec,
    const float* __restrict__ b_dec, const float* __restrict__ acts,
    const float* __restrict__ idxf, const int* __restrict__ eids,
    const float* __restrict__ sel_probs, float* __restrict__ sae_out,
    float* __restrict__ l2)
{
    __shared__ float sa[KTOP];
    __shared__ int   si[KTOP];
    __shared__ float red[256];
    const int t = blockIdx.x, tid = threadIdx.x;
    const int e = eids[t];
    const float p = sel_probs[t];
    if (tid < KTOP) {
        sa[tid] = acts[t * KTOP + tid];
        si[tid] = (int)idxf[t * KTOP + tid];
    }
    __syncthreads();

    float a0 = 0.f, a1 = 0.f, a2 = 0.f;
    const float* Wd = W_dec + (size_t)e * LDIM * DDIM;
    for (int k = 0; k < KTOP; ++k) {
        const float* rp = Wd + (size_t)si[k] * DDIM;
        const float a = sa[k];
        a0 += a * rp[tid]; a1 += a * rp[tid + 256]; a2 += a * rp[tid + 512];
    }
    const float* bd = b_dec + e * DDIM;
    const float* xr = x + (size_t)t * DDIM;
    const float r0 = a0 + bd[tid], r1 = a1 + bd[tid + 256], r2 = a2 + bd[tid + 512];
    float* so = sae_out + (size_t)t * DDIM;
    so[tid] = r0 * p; so[tid + 256] = r1 * p; so[tid + 512] = r2 * p;

    const float d0 = xr[tid] - r0, d1 = xr[tid + 256] - r1, d2 = xr[tid + 512] - r2;
    red[tid] = d0 * d0 + d1 * d1 + d2 * d2;
    __syncthreads();
    for (int s = 128; s > 0; s >>= 1) {
        if (tid < s) red[tid] += red[tid + s];
        __syncthreads();
    }
    if (tid == 0) atomicAdd(&l2[e], red[0]);
}

// ------------------------------------------------------------ finalize
__global__ void finalize_kernel(const int* __restrict__ counts,
                                const float* __restrict__ sumsq,
                                const float* __restrict__ sum_x,
                                const float* __restrict__ l2,
                                float* __restrict__ fvu_o,
                                float* __restrict__ auxk_o,
                                float* __restrict__ multi_o,
                                float* __restrict__ lb_o)
{
    if (threadIdx.x != 0 || blockIdx.x != 0) return;
    float lb = 0.f, fvu = 0.f;
    for (int e = 0; e < NEXP; ++e) {
        const float n = (float)counts[e];
        const float f = n / (float)BTOK - 1.0f / (float)NEXP;
        lb += f * f;
        if (n > 0.f) {
            double s2 = 0.0;
            for (int d = 0; d < DDIM; ++d) {
                const float sx = sum_x[e * DDIM + d];
                s2 += (double)sx * (double)sx;
            }
            const float var = sumsq[e] - (float)(s2 / (double)n);
            if (var > 0.f) fvu += l2[e] / fmaxf(var, 1e-12f) * n;
        }
    }
    fvu_o[0] = fvu / (float)BTOK;
    auxk_o[0] = 0.f;
    multi_o[0] = 0.f;
    lb_o[0] = lb;
}

// -------------------------------------------------------------- launch
extern "C" void kernel_launch(void* const* d_in, const int* in_sizes, int n_in,
                              void* d_out, int out_size, void* d_ws, size_t ws_size,
                              hipStream_t stream)
{
    const float* x        = (const float*)d_in[0];
    const float* router_w = (const float*)d_in[1];
    const float* router_b = (const float*)d_in[2];
    const float* W_enc    = (const float*)d_in[3];
    const float* b_enc    = (const float*)d_in[4];
    const float* W_dec    = (const float*)d_in[5];
    const float* b_dec    = (const float*)d_in[6];

    float* out = (float*)d_out;
    float* sae_o   = out;                                   // [B,D]
    float* acts_o  = out + (size_t)BTOK * DDIM;             // [B,K]
    float* idx_o   = acts_o + (size_t)BTOK * KTOP;          // [B,K]
    float* fvu_o   = idx_o + (size_t)BTOK * KTOP;           // scalar
    float* auxk_o  = fvu_o + 1;                             // scalar
    float* multi_o = auxk_o + 1;                            // scalar
    float* eid_o   = multi_o + 1;                           // [B]
    float* lb_o    = eid_o + BTOK;                          // scalar

    float* wsf       = (float*)d_ws;
    int*   counts    = (int*)(wsf + 0);      // 8
    int*   offsets   = (int*)(wsf + 8);      // 9
    int*   cursor    = (int*)(wsf + 17);     // 8
    float* sumsq     = wsf + 25;             // 8
    float* l2        = wsf + 33;             // 8
    float* sum_x     = wsf + 64;             // 8*768 -> ends at 6208
    float* sel_probs = wsf + 6208;           // 4096
    int*   eids      = (int*)(wsf + 10304);  // 4096
    int*   tok_list  = (int*)(wsf + 14400);  // 4096
    float* pre       = wsf + 18496;          // B*L floats (48 MB)
    __bf16* wencb    = (__bf16*)(wsf + 18496 + (size_t)BTOK * LDIM); // E*L*D bf16 (37.7 MB)

    // zero all accumulators (counts/offsets/cursor/sumsq/l2/sum_x)
    hipMemsetAsync(d_ws, 0, 6208 * sizeof(float), stream);

    // one-shot weight down-conversion: E*L*D / (256*8) blocks
    convert_wenc_kernel<<<(NEXP * LDIM * DDIM) / (256 * 8), 256, 0, stream>>>(
        W_enc, wencb);

    router_kernel<<<BTOK / 8, 256, 0, stream>>>(
        x, router_w, router_b, counts, sumsq, sum_x, sel_probs, eids, eid_o);

    offsets_kernel<<<1, 32, 0, stream>>>(counts, offsets, cursor);

    scatter_kernel<<<BTOK / 256, 256, 0, stream>>>(eids, cursor, tok_list);

    encode_kernel<<<dim3(LDIM / NLAT, NEXP * MTMAX), 128, 0, stream>>>(
        x, wencb, b_enc, b_dec, offsets, tok_list, pre);

    topk_kernel<<<BTOK, 256, 0, stream>>>(pre, acts_o, idx_o);

    decode_kernel<<<BTOK, 256, 0, stream>>>(
        x, W_dec, b_dec, acts_o, idx_o, eids, sel_probs, sae_o, l2);

    finalize_kernel<<<1, 32, 0, stream>>>(
        counts, sumsq, sum_x, l2, fvu_o, auxk_o, multi_o, lb_o);
}
